// MPNN_34402688041445
// MI455X (gfx1250) — compile-verified, matrix operand order checked by reading, not run on hardware
//
#include <hip/hip_runtime.h>

#define N_NODES 25000
#define IN_DIM 16
#define EDGE_DIM 8
#define HID 32
#define OUT_DIM 16
#define N_EDGES 400000
#define MLP_HID 32
#define N_TILES (N_EDGES / 16)   // 25000 tiles of 16 edges
#define N_PAIRS (N_TILES / 2)    // 12500 pairs, no tail

typedef __attribute__((ext_vector_type(16))) _Float16 v16h;
typedef __attribute__((ext_vector_type(8)))  float    v8f;
typedef __attribute__((ext_vector_type(2)))  _Float16 h2;

union AFrag {
  v16h v;
  h2   p[8];   // p[j] = halves {2j, 2j+1}
};

__device__ __forceinline__ v8f wmma16x16x32(v16h a, v16h b, v8f c) {
  // D = A(16x32 f16) * B(32x16 f16) + C(16x16 f32)
  return __builtin_amdgcn_wmma_f32_16x16x32_f16(
      /*neg_a=*/false, a, /*neg_b=*/false, b,
      /*c_mod=*/(short)0, c, /*reuse_a=*/false, /*reuse_b=*/false);
}

__device__ __forceinline__ v16h vzero16h() { v16h z = {}; return z; }

// ---------------------------------------------------------------------------
// Zero accumulation workspace (agg[25000*32] ++ cnt[25000], contiguous).
// ---------------------------------------------------------------------------
__global__ void zero_ws_kernel(float* __restrict__ p, int n) {
  int i = blockIdx.x * blockDim.x + threadIdx.x;
  if (i < n) p[i] = 0.0f;
}

// ---------------------------------------------------------------------------
// Edge kernel. Per wave iteration: a PAIR of 16-edge tiles (32 edges) so all
// B fragments (ds_load) are shared by two A operands:
//   h  = relu(EA @ W1 + b1)            -> 4 WMMA (2 tiles x 2 N-halves)
//   msg = Z @ W2' (+ Xsrc @ b2r step)  -> (16+1) K-steps x 2 N x 2 tiles
//   atomic scatter-add into agg[dst], counts into cnt[dst]
//
// sB: pre-swizzled B fragments, 18 steps (0..15: W2', 16: b2r, 17: W1 pad).
// Fragment K-map (symmetric to documented 16-bit A/C layouts):
//   K(j,lane) = (j&7) + 16*(j>>3) + 8*(lane>>4), col/row = lane&15
// ---------------------------------------------------------------------------
__global__ __launch_bounds__(256) void edge_msg_kernel(
    const float* __restrict__ x, const float* __restrict__ edge_attr,
    const float* __restrict__ W1, const float* __restrict__ b1,
    const float* __restrict__ W2, const float* __restrict__ b2,
    const int* __restrict__ eidx, float* __restrict__ agg,
    float* __restrict__ cnt) {
  __shared__ __align__(32) _Float16 sB[18 * 2 * 32 * 16];   // 36 KB
  __shared__ __align__(8)  _Float16 sH[8 * 2 * 16 * 34];    // f16 h tiles, 17 KB

  const int tid  = threadIdx.x;
  const int lane = tid & 31;
  const int wave = tid >> 5;

  // ---- one-time per-block: build swizzled B-fragment table in LDS ----
  for (int idx = tid; idx < 18 * 2 * 32 * 16; idx += 256) {
    int j  = idx & 15;
    int ln = (idx >> 4) & 31;
    int nh = (idx >> 9) & 1;
    int s  = idx >> 10;
    int K  = (j & 7) + 16 * (j >> 3) + 8 * (ln >> 4);
    int o  = (ln & 15) + 16 * nh;
    float v;
    if (s < 16) {
      int Kg = 32 * s + K;
      v = W2[(Kg >> 4) * (IN_DIM * HID) + (Kg & 15) * HID + o];
    } else if (s == 16) {
      v = (K < IN_DIM) ? b2[K * HID + o] : 0.0f;
    } else {
      v = (K < EDGE_DIM) ? W1[K * MLP_HID + o] : 0.0f;
    }
    sB[idx] = (_Float16)v;
  }
  __syncthreads();

  const int row   = lane & 15;   // A row / C column owned by this lane
  const int hsel  = lane >> 4;
  const int rbase = 8 * hsel;    // C rows owned: rbase .. rbase+7
  const float b1lo = b1[row];
  const float b1hi = b1[16 + row];

  _Float16* hTa = sH + wave * (2 * 16 * 34);
  _Float16* hTb = hTa + 16 * 34;
  const int* srcp = eidx;
  const int* dstp = eidx + N_EDGES;

  for (int pr = blockIdx.x * 8 + wave; pr < N_PAIRS; pr += gridDim.x * 8) {
    const int tA = pr * 2, tB = pr * 2 + 1;

    // ---- gather x_src (this lane: features 8*hsel .. 8*hsel+7) ----
    const int siA = srcp[tA * 16 + row];
    const int siB = srcp[tB * 16 + row];
    const float4* xpA = (const float4*)(x + (size_t)siA * IN_DIM + 8 * hsel);
    const float4* xpB = (const float4*)(x + (size_t)siB * IN_DIM + 8 * hsel);
    float4 a0 = xpA[0], a1 = xpA[1], b0 = xpB[0], b1v = xpB[1];
    h2 xA[4], xB[4];
    xA[0] = (h2){(_Float16)a0.x, (_Float16)a0.y};
    xA[1] = (h2){(_Float16)a0.z, (_Float16)a0.w};
    xA[2] = (h2){(_Float16)a1.x, (_Float16)a1.y};
    xA[3] = (h2){(_Float16)a1.z, (_Float16)a1.w};
    xB[0] = (h2){(_Float16)b0.x, (_Float16)b0.y};
    xB[1] = (h2){(_Float16)b0.z, (_Float16)b0.w};
    xB[2] = (h2){(_Float16)b1v.x, (_Float16)b1v.y};
    xB[3] = (h2){(_Float16)b1v.z, (_Float16)b1v.w};

    // ---- edge-attr A fragments (lanes<16 carry K=j<8, rest zero) ----
    AFrag aEA, aEB;
    aEA.v = vzero16h();
    aEB.v = vzero16h();
    if (lane < 16) {
      const float4* ea = (const float4*)(edge_attr + (size_t)(tA * 16 + row) * EDGE_DIM);
      const float4* eb = (const float4*)(edge_attr + (size_t)(tB * 16 + row) * EDGE_DIM);
      float4 u0 = ea[0], u1 = ea[1], w0 = eb[0], w1 = eb[1];
      aEA.p[0] = (h2){(_Float16)u0.x, (_Float16)u0.y};
      aEA.p[1] = (h2){(_Float16)u0.z, (_Float16)u0.w};
      aEA.p[2] = (h2){(_Float16)u1.x, (_Float16)u1.y};
      aEA.p[3] = (h2){(_Float16)u1.z, (_Float16)u1.w};
      aEB.p[0] = (h2){(_Float16)w0.x, (_Float16)w0.y};
      aEB.p[1] = (h2){(_Float16)w0.z, (_Float16)w0.w};
      aEB.p[2] = (h2){(_Float16)w1.x, (_Float16)w1.y};
      aEB.p[3] = (h2){(_Float16)w1.z, (_Float16)w1.w};
    }

    // ---- h = relu(EA @ W1 + b1) : 4 WMMAs sharing W1 fragments ----
    const v16h bW1lo = *(const v16h*)(sB + ((17 * 2 + 0) * 32 + lane) * 16);
    const v16h bW1hi = *(const v16h*)(sB + ((17 * 2 + 1) * 32 + lane) * 16);
    v8f hA0, hA1, hB0, hB1;
#pragma unroll
    for (int r = 0; r < 8; ++r) {
      hA0[r] = b1lo; hA1[r] = b1hi; hB0[r] = b1lo; hB1[r] = b1hi;
    }
    hA0 = wmma16x16x32(aEA.v, bW1lo, hA0);
    hB0 = wmma16x16x32(aEB.v, bW1lo, hB0);
    hA1 = wmma16x16x32(aEA.v, bW1hi, hA1);
    hB1 = wmma16x16x32(aEB.v, bW1hi, hB1);

    // ---- relu + f16 transpose through per-wave LDS (stride 34 halves) ----
#pragma unroll
    for (int r = 0; r < 8; ++r) {
      hTa[(rbase + r) * 34 + row]      = (_Float16)fmaxf(hA0[r], 0.0f);
      hTa[(rbase + r) * 34 + 16 + row] = (_Float16)fmaxf(hA1[r], 0.0f);
      hTb[(rbase + r) * 34 + row]      = (_Float16)fmaxf(hB0[r], 0.0f);
      hTb[(rbase + r) * 34 + 16 + row] = (_Float16)fmaxf(hB1[r], 0.0f);
    }
    // wave-local store->load ordering (wave32 lockstep, DS in-order per wave)
    asm volatile("s_wait_dscnt 0" ::: "memory");

    // ---- main GEMM: msg = Z @ W2', B fragments shared by both tiles ----
    v8f cA0 = {}, cA1 = {}, cB0 = {}, cB1 = {};
#pragma unroll 4
    for (int s = 0; s < 16; ++s) {
      h2 hpA = *(const h2*)(hTa + row * 34 + 2 * s);  // {h[row,2s], h[row,2s+1]}
      h2 hpB = *(const h2*)(hTb + row * 34 + 2 * s);
      h2 hAl = __builtin_shufflevector(hpA, hpA, 0, 0);
      h2 hAh = __builtin_shufflevector(hpA, hpA, 1, 1);
      h2 hBl = __builtin_shufflevector(hpB, hpB, 0, 0);
      h2 hBh = __builtin_shufflevector(hpB, hpB, 1, 1);
      AFrag fA, fB;
#pragma unroll
      for (int j = 0; j < 4; ++j) {
        fA.p[j]     = hAl * xA[j];   // K hi-bit 0 -> k = 2s
        fA.p[4 + j] = hAh * xA[j];   // K hi-bit 1 -> k = 2s+1
        fB.p[j]     = hBl * xB[j];
        fB.p[4 + j] = hBh * xB[j];
      }
      v16h bb0 = *(const v16h*)(sB + ((s * 2 + 0) * 32 + lane) * 16);
      v16h bb1 = *(const v16h*)(sB + ((s * 2 + 1) * 32 + lane) * 16);
      cA0 = wmma16x16x32(fA.v, bb0, cA0);
      cB0 = wmma16x16x32(fB.v, bb0, cB0);
      cA1 = wmma16x16x32(fA.v, bb1, cA1);
      cB1 = wmma16x16x32(fB.v, bb1, cB1);
    }
    {  // b2 bias K-step: A = [x, 0]
      const v16h bBlo = *(const v16h*)(sB + ((16 * 2 + 0) * 32 + lane) * 16);
      const v16h bBhi = *(const v16h*)(sB + ((16 * 2 + 1) * 32 + lane) * 16);
      AFrag fA, fB;
      fA.v = vzero16h();
      fB.v = vzero16h();
#pragma unroll
      for (int j = 0; j < 4; ++j) { fA.p[j] = xA[j]; fB.p[j] = xB[j]; }
      cA0 = wmma16x16x32(fA.v, bBlo, cA0);
      cB0 = wmma16x16x32(fB.v, bBlo, cB0);
      cA1 = wmma16x16x32(fA.v, bBhi, cA1);
      cB1 = wmma16x16x32(fB.v, bBhi, cB1);
    }

    // ---- scatter-mean accumulation (C row m = r + rbase, column = row) ----
#pragma unroll
    for (int r = 0; r < 8; ++r) {
      int eA = tA * 16 + rbase + r;
      int dA = dstp[eA];
      atomicAdd(&agg[(size_t)dA * HID + row],      cA0[r]);
      atomicAdd(&agg[(size_t)dA * HID + 16 + row], cA1[r]);
      int eB = tB * 16 + rbase + r;
      int dB = dstp[eB];
      atomicAdd(&agg[(size_t)dB * HID + row],      cB0[r]);
      atomicAdd(&agg[(size_t)dB * HID + 16 + row], cB1[r]);
      if (row == 0) {
        atomicAdd(&cnt[dA], 1.0f);
        atomicAdd(&cnt[dB], 1.0f);
      }
    }
  }
}

// ---------------------------------------------------------------------------
// Node kernel: out = relu(agg/max(cnt,1) + x@root + bias) @ Wlin + blin
// ---------------------------------------------------------------------------
__global__ __launch_bounds__(256) void node_out_kernel(
    const float* __restrict__ x, const float* __restrict__ root,
    const float* __restrict__ bias, const float* __restrict__ Wlin,
    const float* __restrict__ blin, const float* __restrict__ agg,
    const float* __restrict__ cnt, float* __restrict__ out) {
  int n = blockIdx.x * blockDim.x + threadIdx.x;
  if (n >= N_NODES) return;
  float c = fmaxf(cnt[n], 1.0f);
  float inv = 1.0f / c;
  float t[HID];
  const float* ag = agg + (size_t)n * HID;
#pragma unroll
  for (int o = 0; o < HID; ++o) t[o] = ag[o] * inv + bias[o];
  const float* xr = x + (size_t)n * IN_DIM;
#pragma unroll 4
  for (int i = 0; i < IN_DIM; ++i) {
    float xi = xr[i];
#pragma unroll
    for (int o = 0; o < HID; ++o) t[o] = fmaf(xi, root[i * HID + o], t[o]);
  }
#pragma unroll
  for (int o = 0; o < HID; ++o) t[o] = fmaxf(t[o], 0.0f);
#pragma unroll 4
  for (int q = 0; q < OUT_DIM; ++q) {
    float acc = blin[q];
#pragma unroll
    for (int k = 0; k < HID; ++k) acc = fmaf(t[k], Wlin[k * OUT_DIM + q], acc);
    out[(size_t)n * OUT_DIM + q] = acc;
  }
}

extern "C" void kernel_launch(void* const* d_in, const int* in_sizes, int n_in,
                              void* d_out, int out_size, void* d_ws,
                              size_t ws_size, hipStream_t stream) {
  (void)in_sizes; (void)n_in; (void)out_size; (void)ws_size;
  const float* x         = (const float*)d_in[0];
  const float* edge_attr = (const float*)d_in[1];
  const float* W1        = (const float*)d_in[2];
  const float* b1        = (const float*)d_in[3];
  const float* W2        = (const float*)d_in[4];
  const float* b2        = (const float*)d_in[5];
  const float* root      = (const float*)d_in[6];
  const float* bias      = (const float*)d_in[7];
  const float* Wlin      = (const float*)d_in[8];
  const float* blin      = (const float*)d_in[9];
  const int*   eidx      = (const int*)d_in[10];

  float* agg = (float*)d_ws;                 // [N_NODES * HID]
  float* cnt = agg + (size_t)N_NODES * HID;  // [N_NODES]
  float* out = (float*)d_out;

  const int zn = N_NODES * (HID + 1);
  zero_ws_kernel<<<(zn + 255) / 256, 256, 0, stream>>>(agg, zn);

  edge_msg_kernel<<<512, 256, 0, stream>>>(x, edge_attr, W1, b1, W2, b2, eidx,
                                           agg, cnt);

  node_out_kernel<<<(N_NODES + 255) / 256, 256, 0, stream>>>(
      x, root, bias, Wlin, blin, agg, cnt, out);
}
